// Block_43817256354477
// MI455X (gfx1250) — matrix operand order repaired and checked by reading, not compiled, not run on hardware
//
#include <hip/hip_runtime.h>
#include <hip/hip_bf16.h>
#include <cstdint>

// ---------- types (plain ext-vectors: trivially-copyable, union-safe) ----------
typedef __attribute__((ext_vector_type(16))) __bf16   v16bf;
typedef __attribute__((ext_vector_type(8)))  float    v8f;
typedef __attribute__((ext_vector_type(4))) unsigned  uint4v;
typedef __attribute__((ext_vector_type(4)))  float    float4v;
typedef __attribute__((ext_vector_type(8)))  int      int8v;
typedef __attribute__((ext_vector_type(4)))  int      int4v;

#if defined(__has_builtin)
#if __has_builtin(__builtin_amdgcn_tensor_load_to_lds) && __has_builtin(__builtin_amdgcn_s_wait_tensorcnt)
#define HAVE_TDM 1
#endif
#endif

static __device__ __forceinline__ unsigned short f2bf(float f) {
  union { float f; unsigned u; } a; a.f = f;
  unsigned u = a.u;
  u += 0x7fffu + ((u >> 16) & 1u);          // round-to-nearest-even
  return (unsigned short)(u >> 16);
}

// =====================================================================
// LayerNorm over last dim (C) : one block per row t
// =====================================================================
__global__ __launch_bounds__(256) void ln_kernel(const float* __restrict__ x,
                                                 float* __restrict__ y,
                                                 const float* __restrict__ w,
                                                 const float* __restrict__ b,
                                                 int C) {
  int t = blockIdx.x;
  const float* xp = x + (size_t)t * C;
  float s = 0.f, s2 = 0.f;
  for (int c = threadIdx.x; c < C; c += 256) { float v = xp[c]; s += v; s2 += v * v; }
  __shared__ float rs[256], rq[256];
  rs[threadIdx.x] = s; rq[threadIdx.x] = s2;
  __syncthreads();
  for (int st = 128; st > 0; st >>= 1) {
    if (threadIdx.x < st) { rs[threadIdx.x] += rs[threadIdx.x + st];
                            rq[threadIdx.x] += rq[threadIdx.x + st]; }
    __syncthreads();
  }
  float mean = rs[0] / C;
  float var  = rq[0] / C - mean * mean;
  float inv  = rsqrtf(var + 1e-5f);
  float* yp = y + (size_t)t * C;
  for (int c = threadIdx.x; c < C; c += 256)
    yp[c] = (xp[c] - mean) * inv * w[c] + b[c];
}

// =====================================================================
// Token-shift mixes
// =====================================================================
__global__ __launch_bounds__(256) void mix4_kernel(const float* __restrict__ x1,
    const float* __restrict__ shift,
    const float* __restrict__ mr, const float* __restrict__ mk,
    const float* __restrict__ mv, const float* __restrict__ mg,
    unsigned short* __restrict__ xr, unsigned short* __restrict__ xk,
    unsigned short* __restrict__ xv, unsigned short* __restrict__ xg,
    int total, int C) {
  int idx = blockIdx.x * 256 + threadIdx.x;
  if (idx >= total) return;
  int c = idx & (C - 1);                    // C is a power of two
  float cur  = x1[idx];
  float prev = (idx < C) ? shift[c] : x1[idx - C];
  float a;
  a = mr[c]; xr[idx] = f2bf(cur * a + prev * (1.f - a));
  a = mk[c]; xk[idx] = f2bf(cur * a + prev * (1.f - a));
  a = mv[c]; xv[idx] = f2bf(cur * a + prev * (1.f - a));
  a = mg[c]; xg[idx] = f2bf(cur * a + prev * (1.f - a));
}

__global__ __launch_bounds__(256) void mix2_kernel(const float* __restrict__ x2,
    const float* __restrict__ shift,
    const float* __restrict__ mk, const float* __restrict__ mr,
    unsigned short* __restrict__ xk, unsigned short* __restrict__ xr,
    int total, int C) {
  int idx = blockIdx.x * 256 + threadIdx.x;
  if (idx >= total) return;
  int c = idx & (C - 1);
  float cur  = x2[idx];
  float prev = (idx < C) ? shift[c] : x2[idx - C];
  float a;
  a = mk[c]; xk[idx] = f2bf(cur * a + prev * (1.f - a));
  a = mr[c]; xr[idx] = f2bf(cur * a + prev * (1.f - a));
}

// =====================================================================
// WKV v5 decay scan: one block per head; thread = (kgroup 0..3, vcol 0..63)
// =====================================================================
__global__ __launch_bounds__(256) void wkv_scan_kernel(const float* __restrict__ r,
    const float* __restrict__ k, const float* __restrict__ v,
    const float* __restrict__ s0, const float* __restrict__ decay,
    const float* __restrict__ faaaa, float* __restrict__ out,
    int T, int C) {
  int h   = blockIdx.x;
  int tid = threadIdx.x;
  int vc  = tid & 63;
  int kg  = tid >> 6;
  float s[16], u[16], w[16];
#pragma unroll
  for (int i = 0; i < 16; i++) {
    int kk = kg * 16 + i;
    s[i] = s0[((size_t)h * 64 + kk) * 64 + vc];
    w[i] = __expf(-__expf(decay[h * 64 + kk]));
    u[i] = faaaa[h * 64 + kk];
  }
  __shared__ float skv[64], svv[64], srr[64];
  __shared__ float red[4][64];
  for (int t = 0; t < T; t++) {
    size_t base = (size_t)t * C + h * 64;
    if (tid < 64)        skv[tid]       = k[base + tid];
    else if (tid < 128)  svv[tid - 64]  = v[base + (tid - 64)];
    else if (tid < 192)  srr[tid - 128] = r[base + (tid - 128)];
    __syncthreads();
    float vv  = svv[vc];
    float acc = 0.f;
#pragma unroll
    for (int i = 0; i < 16; i++) {
      int kk  = kg * 16 + i;
      float a = skv[kk] * vv;
      acc  = fmaf(srr[kk], fmaf(u[i], a, s[i]), acc);
      s[i] = fmaf(s[i], w[i], a);
    }
    red[kg][vc] = acc;
    __syncthreads();
    if (kg == 0)
      out[base + vc] = red[0][vc] + red[1][vc] + red[2][vc] + red[3][vc];
  }
}

// =====================================================================
// GroupNorm(H) on out/8, affine, gate by g, emit bf16 A-operand for Wo
// =====================================================================
__global__ __launch_bounds__(256) void gn_gate_kernel(const float* __restrict__ o,
    const float* __restrict__ g, const float* __restrict__ gw,
    const float* __restrict__ gb, unsigned short* __restrict__ Ao,
    int TH, int H, int C) {
  int idx = blockIdx.x * 256 + threadIdx.x;
  if (idx >= TH) return;
  int t = idx / H, h = idx % H;
  const float* z = o + (size_t)t * C + h * 64;
  float vals[64];
  float m = 0.f;
#pragma unroll
  for (int i = 0; i < 64; i++) { vals[i] = z[i] * 0.125f; m += vals[i]; }
  m *= (1.f / 64.f);
  float var = 0.f;
#pragma unroll
  for (int i = 0; i < 64; i++) { float d = vals[i] - m; var += d * d; }
  var *= (1.f / 64.f);
  float inv = rsqrtf(var + 1e-5f);
#pragma unroll
  for (int i = 0; i < 64; i++) {
    int c = h * 64 + i;
    float zz = (vals[i] - m) * inv * gw[c] + gb[c];
    Ao[(size_t)t * C + c] = f2bf(zz * g[(size_t)t * C + c]);
  }
}

// =====================================================================
// bf16 WMMA GEMM: Y[M,N] = A[M,K](bf16) @ W[N,K]^T (fp32, converted on the fly)
// Block tile 64x256, 8 waves (2x4), wave tile 32x64 (2x4 WMMA frags).
// A tile staged by the Tensor Data Mover (tensor_load_to_lds) with LDS pad
// 16B per 64B row -> 80B padded rows matching the fragment ds_read_b128s.
// EP: 0=f32  1=relu^2->bf16  2=sigmoid  3=silu  4=y+R  5=R+S*y
// =====================================================================
template <int EP>
__global__ __launch_bounds__(256) void gemm_bf16_wmma(
    const unsigned short* __restrict__ A,
    const float* __restrict__ W,
    void* __restrict__ Yv,
    const float* __restrict__ R,
    const float* __restrict__ S,
    int M, int N, int K) {
  constexpr int BM = 64, BN = 256, BK = 32, LDA = 40, LDW = 40; // 80B rows: 16B-aligned, bank-spread
  __shared__ unsigned short sA[BM * LDA];
  __shared__ unsigned short sW[BN * LDW];
  const int tid  = threadIdx.x;
  const int lane = tid & 31;
  const int wid  = tid >> 5;
  const int wm   = wid & 1;            // 2 waves along M
  const int wn   = wid >> 1;           // 4 waves along N
  const int bm   = blockIdx.y * BM;
  const int bn   = blockIdx.x * BN;

  const v8f vzero = {0.f, 0.f, 0.f, 0.f, 0.f, 0.f, 0.f, 0.f};
  v8f acc[2][4];
#pragma unroll
  for (int mi = 0; mi < 2; mi++)
#pragma unroll
    for (int ni = 0; ni < 4; ni++) acc[mi][ni] = vzero;

  const int l16   = lane & 15;
  const int base0 = (lane < 16) ? 0 : 8;   // A-frag K-halves per ISA layout
  const int kb    = (lane < 16) ? 0 : 8;   // B-frag K-base per ISA layout
  union FragU { uint4v q[2]; v16bf v; };

#if defined(HAVE_TDM)
  const unsigned ldsA = (unsigned)(uintptr_t)(&sA[0]);  // LDS byte offset = addr[31:0]
  const unsigned long long gaBase =
      (unsigned long long)(uintptr_t)A + (unsigned long long)bm * (unsigned)K * 2ull;
#endif

  for (int k0 = 0; k0 < K; k0 += BK) {
    __syncthreads();
#if defined(HAVE_TDM)
    if (wid == 0) {  // wave-uniform TDM descriptor: 64 rows x 64B, pad 16B/row
      unsigned long long ga = gaBase + (unsigned long long)k0 * 2ull;
      uint4v g0;
      g0[0] = 1u;                                            // count=1, user mode
      g0[1] = ldsA;                                          // lds_addr
      g0[2] = (unsigned)(ga & 0xffffffffull);                // global_addr lo
      g0[3] = (unsigned)((ga >> 32) & 0x01ffffffull) | (2u << 30);  // addr hi | type=2
      int8v g1;
      g1[0] = (int)((2u << 16)      // data_size = 4B
                  | (1u << 20)      // pad_enable
                  | (3u << 22)      // pad_interval: 16 DW
                  | (3u << 25));    // pad_amount: 4 DW (16B)
      g1[1] = (int)(16u << 16);     // tensor_dim0 lo16 = 16 DW
      g1[2] = (int)(64u << 16);     // tensor_dim0 hi=0 | tensor_dim1 lo16 = 64
      g1[3] = (int)(16u << 16);     // tensor_dim1 hi=0 | tile_dim0 = 16 DW
      g1[4] = 64;                   // tile_dim1 = 64, tile_dim2 = 0
      g1[5] = (int)((unsigned)K >> 1);  // tensor_dim0_stride (DW) lo32
      g1[6] = 0;
      g1[7] = 0;
      int4v gz4 = {0, 0, 0, 0};
      int8v gz8 = {0, 0, 0, 0, 0, 0, 0, 0};
      // 6-arg toolchain variant: (g0, g1, g2, g3, extra, cpol); 2-D tensor -> groups 2/3 unused
      __builtin_amdgcn_tensor_load_to_lds(g0, g1, gz4, gz4, gz8, 0);
    }
#else
    { // manual A stage: 256 threads x 8 bf16
      int aR = tid >> 2, aC = (tid & 3) * 8;
      uint4v d = *reinterpret_cast<const uint4v*>(A + (size_t)(bm + aR) * K + k0 + aC);
      *reinterpret_cast<uint4v*>(&sA[aR * LDA + aC]) = d;
    }
#endif
    { // stage W tile, fp32 -> bf16 (one 32-float row per thread); overlaps the A DMA
      const float* src = W + (size_t)(bn + tid) * K + k0;
      alignas(16) unsigned short tmp[32];
#pragma unroll
      for (int i = 0; i < 32; i += 4) {
        float4v f = *reinterpret_cast<const float4v*>(src + i);
        tmp[i]     = f2bf(f.x); tmp[i + 1] = f2bf(f.y);
        tmp[i + 2] = f2bf(f.z); tmp[i + 3] = f2bf(f.w);
      }
#pragma unroll
      for (int i = 0; i < 4; i++)
        *reinterpret_cast<uint4v*>(&sW[tid * LDW + i * 8]) =
            *reinterpret_cast<const uint4v*>(&tmp[i * 8]);
      if (k0 + BK < K) __builtin_prefetch(src + BK, 0, 1);   // global_prefetch_b8, next K-chunk
    }
#if defined(HAVE_TDM)
    __builtin_amdgcn_s_wait_tensorcnt(0);
#endif
    __syncthreads();

    v16bf afrag[2], bfrag[4];
#pragma unroll
    for (int mi = 0; mi < 2; mi++) {     // A: lane row M=l16, K in {b0..b0+7, b0+16..b0+23}
      const unsigned short* p = &sA[(wm * 32 + mi * 16 + l16) * LDA];
      FragU fu;
      fu.q[0] = *reinterpret_cast<const uint4v*>(p + base0);
      fu.q[1] = *reinterpret_cast<const uint4v*>(p + base0 + 16);
      afrag[mi] = fu.v;
    }
#pragma unroll
    for (int ni = 0; ni < 4; ni++) {     // B: lane col N=l16, K in {kb..kb+7, kb+16..kb+23}
      const unsigned short* p = &sW[(wn * 64 + ni * 16 + l16) * LDW + kb];
      FragU fu;
      fu.q[0] = *reinterpret_cast<const uint4v*>(p);
      fu.q[1] = *reinterpret_cast<const uint4v*>(p + 16);
      bfrag[ni] = fu.v;
    }
#pragma unroll
    for (int mi = 0; mi < 2; mi++)
#pragma unroll
      for (int ni = 0; ni < 4; ni++)
        acc[mi][ni] = __builtin_amdgcn_wmma_f32_16x16x32_bf16(
            false, afrag[mi], false, bfrag[ni], (short)0, acc[mi][ni], false, false);
  }

  // D layout: lane = N + 16*(M>=8); VGPR r -> M = rbase + r
  const int rbase = (lane < 16) ? 0 : 8;
  const size_t idx0 = (size_t)(bm + wm * 32 + rbase) * N + (bn + wn * 64 + l16);
#pragma unroll
  for (int mi = 0; mi < 2; mi++) {
#pragma unroll
    for (int ni = 0; ni < 4; ni++) {
      size_t idx = idx0 + (size_t)(mi * 16) * N + ni * 16;
#pragma unroll
      for (int rr = 0; rr < 8; rr++, idx += N) {
        float y = acc[mi][ni][rr];
        if constexpr (EP == 0) {
          ((float*)Yv)[idx] = y;
        } else if constexpr (EP == 1) {
          float t = y > 0.f ? y : 0.f;
          ((unsigned short*)Yv)[idx] = f2bf(t * t);
        } else if constexpr (EP == 2) {
          ((float*)Yv)[idx] = 1.f / (1.f + __expf(-y));
        } else if constexpr (EP == 3) {
          ((float*)Yv)[idx] = y / (1.f + __expf(-y));
        } else if constexpr (EP == 4) {
          ((float*)Yv)[idx] = y + R[idx];
        } else {
          ((float*)Yv)[idx] = R[idx] + S[idx] * y;
        }
      }
    }
  }
}

// =====================================================================
// Host orchestration
// =====================================================================
extern "C" void kernel_launch(void* const* d_in, const int* in_sizes, int n_in,
                              void* d_out, int out_size, void* d_ws, size_t ws_size,
                              hipStream_t stream) {
  (void)in_sizes; (void)n_in; (void)out_size; (void)ws_size;
  const int T = 1024, C = 2048, H = 32, FF = 7168;

  const float* x        = (const float*)d_in[0];
  const float* attshift = (const float*)d_in[1];
  const float* wkvstate = (const float*)d_in[2];
  const float* ffnshift = (const float*)d_in[3];
  const float* ln1w = (const float*)d_in[4];
  const float* ln1b = (const float*)d_in[5];
  const float* ln2w = (const float*)d_in[6];
  const float* ln2b = (const float*)d_in[7];
  const float* tmk  = (const float*)d_in[8];
  const float* tmv  = (const float*)d_in[9];
  const float* tmr  = (const float*)d_in[10];
  const float* tmg  = (const float*)d_in[11];
  const float* tdec = (const float*)d_in[12];
  const float* tfaa = (const float*)d_in[13];
  const float* Wr   = (const float*)d_in[14];
  const float* Wk   = (const float*)d_in[15];
  const float* Wv   = (const float*)d_in[16];
  const float* Wo   = (const float*)d_in[17];
  const float* Wg   = (const float*)d_in[18];
  const float* gnw  = (const float*)d_in[19];
  const float* gnb  = (const float*)d_in[20];
  const float* cmk  = (const float*)d_in[21];
  const float* cmr  = (const float*)d_in[22];
  const float* Wck  = (const float*)d_in[23];
  const float* Wcr  = (const float*)d_in[24];
  const float* Wcv  = (const float*)d_in[25];

  char* ws = (char*)d_ws;
  size_t off = 0;
  auto alloc = [&](size_t bytes) -> void* {
    void* p = ws + off;
    off += (bytes + 255) & ~(size_t)255;
    return p;
  };
  float*          x1    = (float*)alloc((size_t)T * C * 4);   // ln output (reused for ln2)
  unsigned short* xr    = (unsigned short*)alloc((size_t)T * C * 2);
  unsigned short* xk    = (unsigned short*)alloc((size_t)T * C * 2);
  unsigned short* xv    = (unsigned short*)alloc((size_t)T * C * 2);
  unsigned short* xg    = (unsigned short*)alloc((size_t)T * C * 2);
  float*          rbuf  = (float*)alloc((size_t)T * C * 4);
  float*          kbuf  = (float*)alloc((size_t)T * C * 4);
  float*          vbuf  = (float*)alloc((size_t)T * C * 4);
  float*          gbuf  = (float*)alloc((size_t)T * C * 4);
  float*          oscan = (float*)alloc((size_t)T * C * 4);
  unsigned short* Ao    = (unsigned short*)alloc((size_t)T * C * 2);
  float*          xA    = (float*)alloc((size_t)T * C * 4);
  // reuse dead buffers for ChannelMix
  unsigned short* xk2 = xr;                       // dead after r/k/v/g GEMMs
  unsigned short* xr2 = xk;
  unsigned short* kk  = (unsigned short*)rbuf;    // [T,FF] bf16 (14MB) fits in rbuf+kbuf (16MB)
  float*          sig = vbuf;                     // dead after scan

  dim3 blk(256);
  const int TC_blocks = (T * C + 255) / 256;
  dim3 gemmCC(C / 256, T / 64);
  dim3 gemmFF(FF / 256, T / 64);

  // ---- TimeMix ----
  ln_kernel<<<T, blk, 0, stream>>>(x, x1, ln1w, ln1b, C);
  mix4_kernel<<<TC_blocks, blk, 0, stream>>>(x1, attshift, tmr, tmk, tmv, tmg,
                                             xr, xk, xv, xg, T * C, C);
  gemm_bf16_wmma<0><<<gemmCC, blk, 0, stream>>>(xr, Wr, rbuf, nullptr, nullptr, T, C, C);
  gemm_bf16_wmma<0><<<gemmCC, blk, 0, stream>>>(xk, Wk, kbuf, nullptr, nullptr, T, C, C);
  gemm_bf16_wmma<0><<<gemmCC, blk, 0, stream>>>(xv, Wv, vbuf, nullptr, nullptr, T, C, C);
  gemm_bf16_wmma<3><<<gemmCC, blk, 0, stream>>>(xg, Wg, gbuf, nullptr, nullptr, T, C, C); // silu
  wkv_scan_kernel<<<H, blk, 0, stream>>>(rbuf, kbuf, vbuf, wkvstate, tdec, tfaa, oscan, T, C);
  gn_gate_kernel<<<(T * H + 255) / 256, blk, 0, stream>>>(oscan, gbuf, gnw, gnb, Ao, T * H, H, C);
  gemm_bf16_wmma<4><<<gemmCC, blk, 0, stream>>>(Ao, Wo, xA, x, nullptr, T, C, C);         // + residual x

  // ---- ChannelMix ----
  ln_kernel<<<T, blk, 0, stream>>>(xA, x1, ln2w, ln2b, C);
  mix2_kernel<<<TC_blocks, blk, 0, stream>>>(x1, ffnshift, cmk, cmr, xk2, xr2, T * C, C);
  gemm_bf16_wmma<1><<<gemmFF, blk, 0, stream>>>(xk2, Wck, kk, nullptr, nullptr, T, FF, C); // relu^2 -> bf16
  gemm_bf16_wmma<2><<<gemmCC, blk, 0, stream>>>(xr2, Wcr, sig, nullptr, nullptr, T, C, C); // sigmoid
  gemm_bf16_wmma<5><<<gemmCC, blk, 0, stream>>>(kk, Wcv, d_out, xA, sig, T, C, FF);        // xA + sig*y
}
